// MyModel_87522843559149
// MI455X (gfx1250) — matrix operand order repaired and checked by reading, not compiled, not run on hardware
//
#include <hip/hip_runtime.h>
#include <hip/hip_bf16.h>

// ---------------------------------------------------------------------------
// GRU (U=3) over B=4096, S=256, F=64, then dense(3->10)+softmax.
//
// Phase 1: xk[s][b][0..8] = x[b][s][:] @ kernel + bias[0]   (WMMA f16->f32)
//          stored as f16, padded row stride 16, layout [S, B, 16] in d_ws.
//          Only cols 0..8 are stored (18 MiB written); x loads nontemporal so
//          the intermediate stays resident in the 192 MB L2 for phase 2.
// Phase 2: per-batch-row sequential GRU scan + dense + softmax (VALU).
// ---------------------------------------------------------------------------

typedef __attribute__((ext_vector_type(16))) _Float16 v16h;
typedef __attribute__((ext_vector_type(8)))  _Float16 v8h;
typedef __attribute__((ext_vector_type(8)))  float    v8f;
typedef __attribute__((ext_vector_type(4)))  float    f4;

#define GRU_B 4096
#define GRU_S 256
#define GRU_F 64
#define GRU_G 9          // 3 * U
#define KP    16         // padded f16 row stride in the xk intermediate
#define TILES_PER_WAVE 4 // amortize B-fragment setup

__device__ __forceinline__ void cvt4(v16h& a, int base, f4 p) {
  a[base + 0] = (_Float16)p.x;
  a[base + 1] = (_Float16)p.y;
  a[base + 2] = (_Float16)p.z;
  a[base + 3] = (_Float16)p.w;
}

// ---------------------------------------------------------------------------
// Phase 1: projection GEMM.  Each wave owns TILES_PER_WAVE tiles of 16
// consecutive rows of the flattened [B*S, F] input (16 consecutive rows share
// one batch index b since S % 16 == 0).
//
// 16-bit A-matrix 16x32 layout (ISA 7.12.2): lane L<16 holds row M=L,
//   elems 0..7 = K0..7, elems 8..15 = K16..23 ; lane L>=16 holds row L-16,
//   elems 0..7 = K8..15, elems 8..15 = K24..31.  => 4 contiguous float4
//   global loads per fragment, converted to f16 in registers. No LDS needed.
// 16-bit B-matrix 32x16 layout (same k order as A): lanes 0..15 hold col N=L,
//   K=0..7,16..23 ; lanes 16..31 hold col N=L-16, K=8..15,24..31.
// ---------------------------------------------------------------------------
__global__ __launch_bounds__(256) void gru_proj_wmma(
    const float* __restrict__ x,      // [B, S, F]
    const float* __restrict__ wk,     // [F, 9]
    const float* __restrict__ bias,   // [2, 9]
    _Float16*    __restrict__ xk)     // [S, B, KP] f16
{
  const int lane = threadIdx.x & 31;
  const int wave = blockIdx.x * (blockDim.x >> 5) + (threadIdx.x >> 5);
  const int n    = lane & 15;   // output column this lane owns in C/D
  const int hsel = lane >> 4;   // which half-wave
  const bool live = (n < GRU_G);
  const int  nc   = live ? n : 0;        // clamped column: load unconditionally,
  const float msk = live ? 1.f : 0.f;    // zero via select (no exec divergence)

  // ---- B fragments (K=0..31 and K=32..63), hoisted once per wave ----
  v16h b0, b1;
  #pragma unroll
  for (int i = 0; i < 16; ++i) {
    const int k = (i < 8) ? (hsel * 8 + i) : (16 + hsel * 8 + (i - 8));
    b0[i] = (_Float16)(wk[k * GRU_G + nc] * msk);
    b1[i] = (_Float16)(wk[(k + 32) * GRU_G + nc] * msk);
  }
  const float bias_n = bias[nc] * msk;

  #pragma unroll 2
  for (int t = 0; t < TILES_PER_WAVE; ++t) {
    const long tile = (long)wave * TILES_PER_WAVE + t;
    const long row0 = tile * 16;                       // flattened b*S + s
    // lane's row base, shifted by half-wave K offset (0 or 8 floats)
    const float* xr = x + (row0 + n) * GRU_F + hsel * 8;

    // A fragment, K = 0..31  (nontemporal: x is streamed once, keep L2 for xk)
    f4 p0 = __builtin_nontemporal_load((const f4*)(xr +  0));
    f4 p1 = __builtin_nontemporal_load((const f4*)(xr +  4));
    f4 p2 = __builtin_nontemporal_load((const f4*)(xr + 16));
    f4 p3 = __builtin_nontemporal_load((const f4*)(xr + 20));
    // A fragment, K = 32..63
    f4 q0 = __builtin_nontemporal_load((const f4*)(xr + 32));
    f4 q1 = __builtin_nontemporal_load((const f4*)(xr + 36));
    f4 q2 = __builtin_nontemporal_load((const f4*)(xr + 48));
    f4 q3 = __builtin_nontemporal_load((const f4*)(xr + 52));

    v16h a0, a1;
    cvt4(a0, 0, p0); cvt4(a0, 4, p1); cvt4(a0,  8, p2); cvt4(a0, 12, p3);
    cvt4(a1, 0, q0); cvt4(a1, 4, q1); cvt4(a1,  8, q2); cvt4(a1, 12, q3);

    v8f c = {};
    // D = A0*B0 + A1*B1   (args: neg_a, A, neg_b, B, c_mod, C, reuse_a, reuse_b)
    c = __builtin_amdgcn_wmma_f32_16x16x32_f16(false, a0, false, b0,
                                               (short)0, c, false, false);
    c = __builtin_amdgcn_wmma_f32_16x16x32_f16(false, a1, false, b1,
                                               (short)0, c, false, false);

    // C/D layout: VGPR r holds row m = hsel*8 + r, column n.
    // Store only the 9 live columns (phase 2 never reads cols 9..15).
    const int bidx = (int)(row0 >> 8);                 // row0 / S
    const int s0   = (int)(row0 & 255) + hsel * 8;
    if (live) {
      #pragma unroll
      for (int r = 0; r < 8; ++r) {
        const size_t o = ((size_t)(s0 + r) * GRU_B + bidx) * KP + n;
        xk[o] = (_Float16)(c[r] + bias_n);
      }
    }
  }
}

__device__ __forceinline__ float sigm_f(float v) {
  return 1.f / (1.f + __expf(-v));
}

// ---------------------------------------------------------------------------
// Phase 2: one thread per batch row.  Reads 18B of f16 xk per step
// (lane-contiguous thanks to the [S, B, KP] layout -> mostly L2 hits),
// runs the 3-unit GRU cell, then dense+softmax.
// ---------------------------------------------------------------------------
__global__ __launch_bounds__(256) void gru_scan(
    const _Float16* __restrict__ xk,    // [S, B, KP]
    const float*    __restrict__ rk,    // [U=3, 9]
    const float*    __restrict__ bias,  // [2, 9]
    const float*    __restrict__ dw,    // [3, 10]
    const float*    __restrict__ db,    // [10]
    float*          __restrict__ out)   // [B, 10]
{
  const int b = blockIdx.x * blockDim.x + threadIdx.x;

  float w[27], rb[9];
  #pragma unroll
  for (int i = 0; i < 27; ++i) w[i] = rk[i];     // uniform -> scalar loads
  #pragma unroll
  for (int i = 0; i < 9;  ++i) rb[i] = bias[9 + i];

  float h0 = 0.f, h1 = 0.f, h2 = 0.f;
  const _Float16* p = xk + (size_t)b * KP;
  const size_t step = (size_t)GRU_B * KP;

  #pragma unroll 4
  for (int s = 0; s < GRU_S; ++s) {
    __builtin_prefetch(p + step, 0, 0);          // global_prefetch_b8
    v8h lo = *(const v8h*)p;                     // cols 0..7
    float xt8 = (float)p[8];                     // col 8
    float xt[9];
    #pragma unroll
    for (int j = 0; j < 8; ++j) xt[j] = (float)lo[j];
    xt[8] = xt8;

    float in9[9];
    #pragma unroll
    for (int nn = 0; nn < 9; ++nn)
      in9[nn] = rb[nn] + h0 * w[nn] + h1 * w[9 + nn] + h2 * w[18 + nn];

    const float z0 = sigm_f(xt[0] + in9[0]);
    const float z1 = sigm_f(xt[1] + in9[1]);
    const float z2 = sigm_f(xt[2] + in9[2]);
    const float r0 = sigm_f(xt[3] + in9[3]);
    const float r1 = sigm_f(xt[4] + in9[4]);
    const float r2 = sigm_f(xt[5] + in9[5]);
    const float g0 = tanhf(xt[6] + r0 * in9[6]);
    const float g1 = tanhf(xt[7] + r1 * in9[7]);
    const float g2 = tanhf(xt[8] + r2 * in9[8]);
    h0 = z0 * h0 + (1.f - z0) * g0;
    h1 = z1 * h1 + (1.f - z1) * g1;
    h2 = z2 * h2 + (1.f - z2) * g2;

    p += step;
  }

  // dense(3 -> 10) + softmax
  float lg[10];
  float mx = -1e30f;
  #pragma unroll
  for (int j = 0; j < 10; ++j) {
    lg[j] = db[j] + h0 * dw[j] + h1 * dw[10 + j] + h2 * dw[20 + j];
    mx = fmaxf(mx, lg[j]);
  }
  float sum = 0.f;
  #pragma unroll
  for (int j = 0; j < 10; ++j) { lg[j] = __expf(lg[j] - mx); sum += lg[j]; }
  const float inv = 1.f / sum;
  #pragma unroll
  for (int j = 0; j < 10; ++j) out[b * 10 + j] = lg[j] * inv;
}

extern "C" void kernel_launch(void* const* d_in, const int* in_sizes, int n_in,
                              void* d_out, int out_size, void* d_ws, size_t ws_size,
                              hipStream_t stream) {
  (void)in_sizes; (void)n_in; (void)out_size; (void)ws_size;
  const float* x    = (const float*)d_in[0];   // [4096, 256, 64]
  const float* wk   = (const float*)d_in[1];   // [64, 9]
  const float* rk   = (const float*)d_in[2];   // [3, 9]
  const float* bias = (const float*)d_in[3];   // [2, 9]
  const float* dw   = (const float*)d_in[4];   // [3, 10]
  const float* db   = (const float*)d_in[5];   // [10]
  float*       out  = (float*)d_out;           // [4096, 10]

  // Workspace: xk intermediate, f16 [S, B, KP] = 256*4096*16*2 = 32 MiB.
  _Float16* xk = (_Float16*)d_ws;

  // Phase 1: 65536 tiles / (8 waves * 4 tiles per wave) = 2048 blocks.
  gru_proj_wmma<<<2048, 256, 0, stream>>>(x, wk, bias, xk);
  // Phase 2: 4096 rows / 256 threads = 16 blocks.
  gru_scan<<<GRU_B / 256, 256, 0, stream>>>(xk, rk, bias, dw, db, out);
}
